// HeteroGraphEncoder_51969104281734
// MI455X (gfx1250) — matrix operand order repaired and checked by reading, not compiled, not run on hardware
//
#include <hip/hip_runtime.h>

// ---------------------------------------------------------------------------
// HeteroGraphEncoder (3-layer GATv2) for gfx1250 (MI455X), wave32.
//  - Dense transforms: v_wmma_f32_16x16x32_f16, weights pre-transposed to
//    f16 [Nc][K] so the B fragment is one aligned 32B v16h load; A fragment
//    is two contiguous 32B float4-pair loads per lane.
//  - Edge phase: wave-per-edge, We/att staged in LDS, L2-resident node
//    tensors, ordered-uint atomicMax for segment max, f32 atomicAdd scatter.
// ---------------------------------------------------------------------------

typedef __attribute__((ext_vector_type(16))) _Float16 v16h;
typedef __attribute__((ext_vector_type(8)))  float    v8f;

#define HIDC   64
#define NHEAD  4
#define DCH    256   // HEADS*HID

// ---- float <-> order-preserving unsigned (for atomicMax on floats) --------
__device__ __forceinline__ unsigned ord_enc(float x) {
  unsigned u = __float_as_uint(x);
  return (u & 0x80000000u) ? ~u : (u | 0x80000000u);
}
__device__ __forceinline__ float ord_dec(unsigned u) {
  u = (u & 0x80000000u) ? (u ^ 0x80000000u) : ~u;
  return __uint_as_float(u);
}

// ---- generic fill ----------------------------------------------------------
__global__ void hg_fill_u32(unsigned* __restrict__ p, unsigned v, size_t n) {
  size_t i = (size_t)blockIdx.x * blockDim.x + threadIdx.x;
  size_t stride = (size_t)gridDim.x * blockDim.x;
  for (; i < n; i += stride) p[i] = v;
}

// ---- one-time weight transpose + f32->f16 convert: Wt[col][k] = W[k][col] --
__global__ void hg_w_t16(const float* __restrict__ W, _Float16* __restrict__ Wt,
                         int K, int Nc) {
  const int t = blockIdx.x * blockDim.x + threadIdx.x;
  if (t >= K * Nc) return;
  const int col = t / K, k = t - col * K;      // t = col*K + k -> coalesced writes
  Wt[(size_t)col * K + k] = (_Float16)W[(size_t)k * Nc + col];
}

// ---- WMMA GEMM: OUT[M x NC] = A[M x K] @ W[K x NC] + bias ------------------
// One 16x16 tile per wave; 4 waves per block cover 4 consecutive N-tiles.
// Fragment layouts used here (wave32; contiguous k order, same in A and B):
//   A (16x32 f16): lane L -> row L&15; a[j] = A[row][k0 + half*16 + j]
//                  -> two contiguous 32B chunks per lane.
//   B (32x16 f16): lane L -> col L&15; b[j] = W[k0+half*16+j][col]
//                  -> one contiguous 32B v16h load from transposed Wt.
//   C/D (16x16 f32): element r -> row (L/16)*8 + r, col L&15
template<int K, int NC>
__launch_bounds__(128)
__global__ void hg_gemm_bias(const float* __restrict__ A, const _Float16* __restrict__ Wt,
                             const float* __restrict__ bias, float* __restrict__ OUT,
                             int M) {
  const int lane = threadIdx.x & 31;
  const int wave = threadIdx.x >> 5;
  const int nt = blockIdx.y * 4 + wave;
  if (nt * 16 >= NC) return;                   // wave-uniform
  const int mt   = blockIdx.x;
  const int half = lane >> 4;
  int arow = mt * 16 + (lane & 15);
  if (arow >= M) arow = M - 1;                 // clamp (harness M is 16-aligned)
  const int col = nt * 16 + (lane & 15);
  const float*    Arow = A  + (size_t)arow * K;
  const _Float16* Bcol = Wt + (size_t)col  * K;

  v8f acc = {};
  #pragma unroll
  for (int k0 = 0; k0 < K; k0 += 32) {
    const float4* Ap0 = (const float4*)(Arow + k0 +     (half << 4));
    const float4* Ap1 = (const float4*)(Arow + k0 + 8 + (half << 4));
    const float4 f0 = Ap0[0], f1 = Ap0[1];     // k = k0+half*16   .. +7
    const float4 f2 = Ap1[0], f3 = Ap1[1];     // k = k0+half*16+8 .. +15
    v16h a;
    a[0]=(_Float16)f0.x;  a[1]=(_Float16)f0.y;  a[2]=(_Float16)f0.z;  a[3]=(_Float16)f0.w;
    a[4]=(_Float16)f1.x;  a[5]=(_Float16)f1.y;  a[6]=(_Float16)f1.z;  a[7]=(_Float16)f1.w;
    a[8]=(_Float16)f2.x;  a[9]=(_Float16)f2.y;  a[10]=(_Float16)f2.z; a[11]=(_Float16)f2.w;
    a[12]=(_Float16)f3.x; a[13]=(_Float16)f3.y; a[14]=(_Float16)f3.z; a[15]=(_Float16)f3.w;
    const v16h b = *(const v16h*)(Bcol + k0 + (half << 4));   // 32B aligned
    acc = __builtin_amdgcn_wmma_f32_16x16x32_f16(
        /*neg_a=*/false, a, /*neg_b=*/false, b,
        /*c_mod=*/(short)0, acc, /*reuse_a=*/false, /*reuse_b=*/false);
  }

  const float bv = bias[col];
  const int rbase = mt * 16 + (half << 3);
  float* Orow = OUT + (size_t)rbase * NC + col;
  if (mt * 16 + 16 <= M) {                     // wave-uniform fast path
    #pragma unroll
    for (int r = 0; r < 8; ++r) Orow[r * NC] = acc[r] + bv;
  } else {
    #pragma unroll
    for (int r = 0; r < 8; ++r)
      if (rbase + r < M) Orow[r * NC] = acc[r] + bv;
  }
}

// ---- edge pass 1: attention logits + segment max ---------------------------
// One wave per edge; lane handles channels c = lane + 32*j (coalesced 128B).
// We (5x256) and att (256) staged in LDS once per block.
__launch_bounds__(256)
__global__ void hg_edge_alpha(const float* __restrict__ xl, const float* __restrict__ xr,
                              const float* __restrict__ edge_attr,
                              const float* __restrict__ We, const float* __restrict__ att,
                              const int* __restrict__ src, const int* __restrict__ dst,
                              const float* __restrict__ temps, int layer,
                              float* __restrict__ alpha, unsigned* __restrict__ amax, int E) {
  __shared__ float sWe[5 * DCH];
  __shared__ float sAtt[DCH];
  for (int t = threadIdx.x; t < 5 * DCH; t += 256) sWe[t] = We[t];
  for (int t = threadIdx.x; t < DCH; t += 256)     sAtt[t] = att[t];
  __syncthreads();                              // before any divergent exit

  const int lane = threadIdx.x & 31;
  const int e = blockIdx.x * 8 + (threadIdx.x >> 5);
  if (e >= E) return;
  const int s = src[e], d = dst[e];

  float ea[5];
  #pragma unroll
  for (int q = 0; q < 5; ++q) ea[q] = edge_attr[(size_t)e * 5 + q];
  ea[4] = temps[layer] / fmaxf(ea[3], 1e-6f);   // physics: temp / clip(z,1e-6)

  float hs0 = 0.f, hs1 = 0.f, hs2 = 0.f, hs3 = 0.f;
  #pragma unroll
  for (int j = 0; j < 8; ++j) {
    const int c = lane + 32 * j;                // head = j>>1 (compile-time)
    float eec = 0.f;
    #pragma unroll
    for (int q = 0; q < 5; ++q) eec += ea[q] * sWe[q * DCH + c];
    float m = xl[(size_t)s * DCH + c] + xr[(size_t)d * DCH + c] + eec;
    m = m > 0.f ? m : 0.2f * m;                 // leaky_relu(0.2)
    const float t = m * sAtt[c];                // att flat [4][64] == flat [256]
    if (j < 2) hs0 += t; else if (j < 4) hs1 += t; else if (j < 6) hs2 += t; else hs3 += t;
  }
  #pragma unroll
  for (int off = 16; off > 0; off >>= 1) {
    hs0 += __shfl_xor(hs0, off, 32);
    hs1 += __shfl_xor(hs1, off, 32);
    hs2 += __shfl_xor(hs2, off, 32);
    hs3 += __shfl_xor(hs3, off, 32);
  }
  if (lane < 4) {
    const float v = lane == 0 ? hs0 : lane == 1 ? hs1 : lane == 2 ? hs2 : hs3;
    alpha[(size_t)e * 4 + lane] = v;
    atomicMax(&amax[(size_t)d * 4 + lane], ord_enc(v));
  }
}

// ---- edge pass 2: exp(alpha - max) and denominator -------------------------
__launch_bounds__(256)
__global__ void hg_edge_exp(float* __restrict__ alpha, const unsigned* __restrict__ amax,
                            const int* __restrict__ dst, float* __restrict__ den, int E) {
  const int t = blockIdx.x * blockDim.x + threadIdx.x;
  if (t >= E * 4) return;
  const int e = t >> 2, hd = t & 3;
  const int d = dst[e];
  const float v = __expf(alpha[t] - ord_dec(amax[(size_t)d * 4 + hd]));
  alpha[t] = v;                                 // overwrite with ex
  atomicAdd(&den[(size_t)d * 4 + hd], v);
}

// ---- edge pass 3: weighted scatter-sum of xl[src] into out[dst] ------------
__launch_bounds__(256)
__global__ void hg_edge_scatter(const float* __restrict__ ex, const float* __restrict__ den,
                                const float* __restrict__ xl,
                                const int* __restrict__ src, const int* __restrict__ dst,
                                float* __restrict__ out, int E) {
  const int lane = threadIdx.x & 31;
  const int e = blockIdx.x * 8 + (threadIdx.x >> 5);
  if (e >= E) return;
  const int s = src[e], d = dst[e];
  float a[4];
  #pragma unroll
  for (int hd = 0; hd < 4; ++hd)
    a[hd] = ex[(size_t)e * 4 + hd] / fmaxf(den[(size_t)d * 4 + hd], 1e-16f);
  #pragma unroll
  for (int j = 0; j < 8; ++j) {
    const int c = lane + 32 * j;                // head = j>>1
    atomicAdd(&out[(size_t)d * DCH + c], a[j >> 1] * xl[(size_t)s * DCH + c]);
  }
}

// ---- node finalize: head-mean/concat + bias + LayerNorm + residual + ELU ---
template<int DOUT>
__launch_bounds__(256)
__global__ void hg_node_finalize(const float* __restrict__ acc, const float* __restrict__ bo,
                                 const float* __restrict__ gam, const float* __restrict__ bet,
                                 const float* __restrict__ resid, float* __restrict__ hout,
                                 int Nn, int concat, int do_elu) {
  const int lane = threadIdx.x & 31;
  const int node = blockIdx.x * 8 + (threadIdx.x >> 5);
  if (node >= Nn) return;
  constexpr int PER = DOUT / 32;
  const float* row = acc + (size_t)node * DCH;

  float v[PER];
  #pragma unroll
  for (int j = 0; j < PER; ++j) {
    const int c = lane + 32 * j;
    float t;
    if (concat) t = row[c];
    else        t = 0.25f * (row[c] + row[c + 64] + row[c + 128] + row[c + 192]);
    v[j] = t + bo[c];
  }
  float s = 0.f;
  #pragma unroll
  for (int j = 0; j < PER; ++j) s += v[j];
  #pragma unroll
  for (int off = 16; off > 0; off >>= 1) s += __shfl_xor(s, off, 32);
  const float mu = s / (float)DOUT;
  float vs = 0.f;
  #pragma unroll
  for (int j = 0; j < PER; ++j) { const float dv = v[j] - mu; vs += dv * dv; }
  #pragma unroll
  for (int off = 16; off > 0; off >>= 1) vs += __shfl_xor(vs, off, 32);
  const float rstd = rsqrtf(vs / (float)DOUT + 1e-5f);

  #pragma unroll
  for (int j = 0; j < PER; ++j) {
    const int c = lane + 32 * j;
    float o = (v[j] - mu) * rstd * gam[c] + bet[c] + resid[(size_t)node * DOUT + c];
    if (do_elu) o = o > 0.f ? o : (__expf(o) - 1.f);
    hout[(size_t)node * DOUT + c] = o;
  }
}

// ---------------------------------------------------------------------------
extern "C" void kernel_launch(void* const* d_in, const int* in_sizes, int n_in,
                              void* d_out, int out_size, void* d_ws, size_t ws_size,
                              hipStream_t stream) {
  (void)n_in; (void)out_size; (void)ws_size;
  const float* x         = (const float*)d_in[0];
  const float* edge_attr = (const float*)d_in[1];
  const float* proj_W    = (const float*)d_in[2];
  const float* proj_b    = (const float*)d_in[3];
  const float* Wl[3]  = {(const float*)d_in[4],  (const float*)d_in[5],  (const float*)d_in[6]};
  const float* bl[3]  = {(const float*)d_in[7],  (const float*)d_in[8],  (const float*)d_in[9]};
  const float* Wr[3]  = {(const float*)d_in[10], (const float*)d_in[11], (const float*)d_in[12]};
  const float* br[3]  = {(const float*)d_in[13], (const float*)d_in[14], (const float*)d_in[15]};
  const float* We[3]  = {(const float*)d_in[16], (const float*)d_in[17], (const float*)d_in[18]};
  const float* att[3] = {(const float*)d_in[19], (const float*)d_in[20], (const float*)d_in[21]};
  const float* bo[3]  = {(const float*)d_in[22], (const float*)d_in[23], (const float*)d_in[24]};
  const float* lg[3]  = {(const float*)d_in[25], (const float*)d_in[26], (const float*)d_in[27]};
  const float* lb[3]  = {(const float*)d_in[28], (const float*)d_in[29], (const float*)d_in[30]};
  const float* temps  = (const float*)d_in[31];
  const float* rW0 = (const float*)d_in[32];
  const float* rb0 = (const float*)d_in[33];
  const float* rW2 = (const float*)d_in[34];
  const float* rb2 = (const float*)d_in[35];
  const int*   eidx = (const int*)d_in[36];

  const int Nn = in_sizes[0] / 32;     // 50000
  const int E  = in_sizes[1] / 5;      // 800000
  const int* src = eidx;
  const int* dst = eidx + E;

  // workspace carve-up
  float* p = (float*)d_ws;
  float* hA    = p; p += (size_t)Nn * DCH;
  float* hB    = p; p += (size_t)Nn * DCH;
  float* xl    = p; p += (size_t)Nn * DCH;
  float* xr    = p; p += (size_t)Nn * DCH;
  float* rp    = p; p += (size_t)Nn * DCH;
  float* oacc  = p; p += (size_t)Nn * DCH;
  float* alpha = p; p += (size_t)E * 4;
  unsigned* amax = (unsigned*)p; p += (size_t)Nn * 4;
  float* den   = p; p += (size_t)Nn * 4;
  // f16 transposed weights (32B-aligned: all prior sizes are 32B multiples)
  _Float16* q = (_Float16*)p;
  _Float16* tProj = q;  q += 32 * 64;
  _Float16* tWl[3]; tWl[0] = q; q += 64 * 256; tWl[1] = q; q += 256 * 256; tWl[2] = q; q += 256 * 256;
  _Float16* tWr[3]; tWr[0] = q; q += 64 * 256; tWr[1] = q; q += 256 * 256; tWr[2] = q; q += 256 * 256;
  _Float16* tR0 = q; q += 64 * 256;
  _Float16* tR2 = q; q += 256 * 64;

  // one-time weight transpose/convert (weights are constant inputs)
  auto wt = [&](const float* W, _Float16* T, int K, int Nc) {
    hg_w_t16<<<(K * Nc + 255) / 256, 256, 0, stream>>>(W, T, K, Nc);
  };
  wt(proj_W, tProj, 32, 64);
  wt(Wl[0], tWl[0], 64, 256);  wt(Wl[1], tWl[1], 256, 256);  wt(Wl[2], tWl[2], 256, 256);
  wt(Wr[0], tWr[0], 64, 256);  wt(Wr[1], tWr[1], 256, 256);  wt(Wr[2], tWr[2], 256, 256);
  wt(rW0, tR0, 64, 256);
  wt(rW2, tR2, 256, 64);

  const int mt = (Nn + 15) / 16;
  const int eb = (E + 7) / 8;          // wave-per-edge, 8 waves/block
  const int nb = (Nn + 7) / 8;

  // input projection: h = x @ proj_W + proj_b   [N,32]@[32,64]
  hg_gemm_bias<32, 64><<<dim3(mt, 1), dim3(128), 0, stream>>>(x, tProj, proj_b, hA, Nn);

  float* hcur = hA; float* hnext = hB;
  const int dins[3]   = {64, 256, 256};
  const int douts[3]  = {256, 256, 64};
  const int concat[3] = {1, 1, 0};

  for (int i = 0; i < 3; ++i) {
    // node transforms (Nc = 256 for Wl/Wr)
    if (dins[i] == 64) {
      hg_gemm_bias<64, 256><<<dim3(mt, 4), dim3(128), 0, stream>>>(hcur, tWl[i], bl[i], xl, Nn);
      hg_gemm_bias<64, 256><<<dim3(mt, 4), dim3(128), 0, stream>>>(hcur, tWr[i], br[i], xr, Nn);
    } else {
      hg_gemm_bias<256, 256><<<dim3(mt, 4), dim3(128), 0, stream>>>(hcur, tWl[i], bl[i], xl, Nn);
      hg_gemm_bias<256, 256><<<dim3(mt, 4), dim3(128), 0, stream>>>(hcur, tWr[i], br[i], xr, Nn);
    }
    // residual (projected on layers 0 and 2)
    const float* resid = hcur;
    if (i == 0) { hg_gemm_bias<64, 256><<<dim3(mt, 4), dim3(128), 0, stream>>>(hcur, tR0, rb0, rp, Nn); resid = rp; }
    if (i == 2) { hg_gemm_bias<256, 64><<<dim3(mt, 1), dim3(128), 0, stream>>>(hcur, tR2, rb2, rp, Nn); resid = rp; }

    // per-layer accumulator init
    hg_fill_u32<<<2048, 256, 0, stream>>>((unsigned*)oacc, 0u, (size_t)Nn * DCH);
    hg_fill_u32<<<64,   256, 0, stream>>>((unsigned*)den,  0u, (size_t)Nn * 4);
    hg_fill_u32<<<64,   256, 0, stream>>>(amax,            0u, (size_t)Nn * 4); // ~ -inf

    // edge phase
    hg_edge_alpha  <<<eb, 256, 0, stream>>>(xl, xr, edge_attr, We[i], att[i],
                                            src, dst, temps, i, alpha, amax, E);
    hg_edge_exp    <<<(E * 4 + 255) / 256, 256, 0, stream>>>(alpha, amax, dst, den, E);
    hg_edge_scatter<<<eb, 256, 0, stream>>>(alpha, den, xl, src, dst, oacc, E);

    // node finalize (layer 2 writes straight to d_out)
    float* hdst = (i == 2) ? (float*)d_out : hnext;
    if (douts[i] == 256)
      hg_node_finalize<256><<<nb, 256, 0, stream>>>(oacc, bo[i], lg[i], lb[i], resid, hdst,
                                                    Nn, concat[i], (i < 2) ? 1 : 0);
    else
      hg_node_finalize<64> <<<nb, 256, 0, stream>>>(oacc, bo[i], lg[i], lb[i], resid, hdst,
                                                    Nn, concat[i], (i < 2) ? 1 : 0);

    float* t2 = hcur; hcur = hnext; hnext = t2;
  }
}
